// IrrepsConvolution_74723841016349
// MI455X (gfx1250) — matrix operand order repaired and checked by reading, not compiled, not run on hardware
//
#include <hip/hip_runtime.h>
#include <cmath>

// ---------------------------------------------------------------------------
// Types for CDNA5 WMMA
// ---------------------------------------------------------------------------
typedef __attribute__((ext_vector_type(16))) _Float16 v16h;
typedef __attribute__((ext_vector_type(8)))  _Float16 v8h;
typedef __attribute__((ext_vector_type(8)))  float    v8f;
typedef __attribute__((ext_vector_type(4)))  float    v4f;

#define N_NODES   20000
#define N_EDGES   640000
#define EMB_DIM   8
#define HID       64
#define W_NUMEL   128          // 4 * MUL
#define MUL       32

#define LOG2F_C   0.6931471805599453f
#define INV_SQRT3 0.57735026918962576f
#define INV_SQRT8 0.35355339059327373f   // 1/sqrt(EMB_DIM)
#define INV_8     0.125f                 // 1/sqrt(HID)

// ws layout (f16 halves): W1t [64 cols][32 K(pad)] | W2t [64][64] | W3t [128][64]
#define W1T_OFF   0
#define W2T_OFF   (64 * 32)
#define W3T_OFF   (64 * 32 + 64 * 64)

// ---------------------------------------------------------------------------
// Kernel 1: transpose + scale + f16-convert the MLP weights into workspace.
// Column-major ("Wt[n*K + k]") so each WMMA B fragment is one contiguous
// 32-byte load per lane.
// ---------------------------------------------------------------------------
__global__ void prep_weights_kernel(const float* __restrict__ W1,
                                    const float* __restrict__ W2,
                                    const float* __restrict__ W3,
                                    unsigned short* __restrict__ ws_raw) {
  _Float16* ws = (_Float16*)ws_raw;
  _Float16* W1t = ws + W1T_OFF;
  _Float16* W2t = ws + W2T_OFF;
  _Float16* W3t = ws + W3T_OFF;
  const int tid    = threadIdx.x + blockIdx.x * blockDim.x;
  const int stride = blockDim.x * gridDim.x;

  for (int i = tid; i < 64 * 32; i += stride) {         // W1: 8x64 -> [64][32], K padded
    int n = i >> 5, k = i & 31;
    float v = (k < EMB_DIM) ? W1[k * 64 + n] * INV_SQRT8 : 0.0f;
    W1t[n * 32 + k] = (_Float16)v;
  }
  for (int i = tid; i < 64 * 64; i += stride) {         // W2: 64x64 -> [64][64]
    int n = i >> 6, k = i & 63;
    W2t[n * 64 + k] = (_Float16)(W2[k * 64 + n] * INV_8);
  }
  for (int i = tid; i < 128 * 64; i += stride) {        // W3: 64x128 -> [128][64]
    int n = i >> 6, k = i & 63;
    W3t[n * 64 + k] = (_Float16)(W3[k * 128 + n] * INV_8);
  }
}

// ---------------------------------------------------------------------------
// shifted softplus: (softplus(x) - log 2) * ssp_c
// Branch-free + overflow-safe: softplus(x) = max(x,0) + log(1 + exp(-|x|)).
// Uses hardware transcendentals (v_exp_f32 / v_log_f32).
// ---------------------------------------------------------------------------
__device__ __forceinline__ float ssp_f(float v, float c) {
  float sp = fmaxf(v, 0.0f) + __logf(1.0f + __expf(-fabsf(v)));
  return (sp - LOG2F_C) * c;
}

__device__ __forceinline__ void atomic_add_f32(float* p, float v) {
  unsafeAtomicAdd(p, v);   // native global_atomic_add_f32
}

// ---------------------------------------------------------------------------
// Kernel 2: fused MLP (WMMA) + tensor-product message + atomic scatter-add.
// 256 threads = 8 waves; each wave handles a 16-edge tile.
// ---------------------------------------------------------------------------
__global__ void __launch_bounds__(256)
edge_mlp_message_kernel(const float* __restrict__ x,
                        const float* __restrict__ edge_attr,
                        const float* __restrict__ edge_emb,
                        const int*   __restrict__ edge_idx,
                        const unsigned short* __restrict__ wts_raw,
                        const float* __restrict__ denom,
                        float* __restrict__ out,
                        float ssp_c) {
  const _Float16* wts = (const _Float16*)wts_raw;
  const _Float16* W1t = wts + W1T_OFF;
  const _Float16* W2t = wts + W2T_OFF;
  const _Float16* W3t = wts + W3T_OFF;

  // 8 KB per wave: first used as f16 activation staging (16x64 halves = 2 KB),
  // later overwritten with the f32 per-edge weights (16x128 = 8 KB).
  __shared__ float smem[8 * 16 * 128];   // 64 KB

  const int lane = threadIdx.x & 31;
  const int wave = threadIdx.x >> 5;
  const int lh   = lane >> 4;            // lane half: 0 or 1
  const int r    = lane & 15;            // row (A M) or col (B/C N) within tile
  const int e0   = blockIdx.x * 128 + wave * 16;   // first edge of this wave

  float*     wb = smem + wave * 16 * 128;
  _Float16*  hb = (_Float16*)wb;         // 16 rows x 64 cols, row-major f16

  // ---- Layer 1: A = edge_emb tile (16x8, K zero-padded to 32) --------------
  v16h a1;
  #pragma unroll
  for (int i = 0; i < 16; ++i) a1[i] = (_Float16)0.0f;
  if (lh == 0) {
    // lane half 0 holds K = 0..7 in halves 0..7 (rest of K range is padding)
    const v4f* er = (const v4f*)(edge_emb + (size_t)(e0 + r) * EMB_DIM);
    v4f e_lo = er[0], e_hi = er[1];
    #pragma unroll
    for (int k = 0; k < 4; ++k) { a1[k] = (_Float16)e_lo[k]; a1[4 + k] = (_Float16)e_hi[k]; }
  }

  v8f h1[4];
  #pragma unroll
  for (int t = 0; t < 4; ++t) {
    v16h b = *(const v16h*)(W1t + (t * 16 + r) * 32 + lh * 16);
    v8f c;
    #pragma unroll
    for (int i = 0; i < 8; ++i) c[i] = 0.0f;
    c = __builtin_amdgcn_wmma_f32_16x16x32_f16(false, a1, false, b, (short)0, c, false, false);
    h1[t] = c;
  }

  // ssp + stage to LDS (row-major f16, D-layout: M = v + 8*lh, N = t*16 + r)
  #pragma unroll
  for (int t = 0; t < 4; ++t)
    #pragma unroll
    for (int v = 0; v < 8; ++v)
      hb[(v + 8 * lh) * 64 + (t * 16 + r)] = (_Float16)ssp_f(h1[t][v], ssp_c);
  __syncthreads();

  // ---- Layer 2: A fragments from LDS (two contiguous 16B loads each) -------
  v16h a2[2];
  #pragma unroll
  for (int kb = 0; kb < 2; ++kb) {
    v8h lo = *(const v8h*)(hb + r * 64 + kb * 32 + lh * 16);
    v8h hi = *(const v8h*)(hb + r * 64 + kb * 32 + lh * 16 + 8);
    #pragma unroll
    for (int i = 0; i < 8; ++i) { a2[kb][i] = lo[i]; a2[kb][8 + i] = hi[i]; }
  }

  v8f h2[4];
  #pragma unroll
  for (int t = 0; t < 4; ++t) {
    v8f c;
    #pragma unroll
    for (int i = 0; i < 8; ++i) c[i] = 0.0f;
    #pragma unroll
    for (int kb = 0; kb < 2; ++kb) {
      v16h b = *(const v16h*)(W2t + (t * 16 + r) * 64 + kb * 32 + lh * 16);
      c = __builtin_amdgcn_wmma_f32_16x16x32_f16(false, a2[kb], false, b, (short)0, c, false, false);
    }
    h2[t] = c;
  }
  __syncthreads();

  #pragma unroll
  for (int t = 0; t < 4; ++t)
    #pragma unroll
    for (int v = 0; v < 8; ++v)
      hb[(v + 8 * lh) * 64 + (t * 16 + r)] = (_Float16)ssp_f(h2[t][v], ssp_c);
  __syncthreads();

  // ---- Layer 3: 16x128 output = per-edge tensor-product weights ------------
  v16h a3[2];
  #pragma unroll
  for (int kb = 0; kb < 2; ++kb) {
    v8h lo = *(const v8h*)(hb + r * 64 + kb * 32 + lh * 16);
    v8h hi = *(const v8h*)(hb + r * 64 + kb * 32 + lh * 16 + 8);
    #pragma unroll
    for (int i = 0; i < 8; ++i) { a3[kb][i] = lo[i]; a3[kb][8 + i] = hi[i]; }
  }

  v8f wacc[8];
  #pragma unroll
  for (int t = 0; t < 8; ++t) {
    v8f c;
    #pragma unroll
    for (int i = 0; i < 8; ++i) c[i] = 0.0f;
    #pragma unroll
    for (int kb = 0; kb < 2; ++kb) {
      v16h b = *(const v16h*)(W3t + (t * 16 + r) * 64 + kb * 32 + lh * 16);
      c = __builtin_amdgcn_wmma_f32_16x16x32_f16(false, a3[kb], false, b, (short)0, c, false, false);
    }
    wacc[t] = c;
  }
  __syncthreads();   // make sure all lanes are done reading hb before overwrite

  #pragma unroll
  for (int t = 0; t < 8; ++t)
    #pragma unroll
    for (int v = 0; v < 8; ++v)
      wb[(v + 8 * lh) * 128 + t * 16 + r] = wacc[t][v];
  __syncthreads();

  // ---- Message construction + atomic scatter-add ---------------------------
  // Lane = channel u (0..31); loop over the wave's 16 edges (wave-uniform e).
  const float inv_den = 1.0f / denom[0];
  const int u = lane;

  for (int el = 0; el < 16; ++el) {
    const int e   = e0 + el;
    const int dst = edge_idx[e];
    const int src = edge_idx[N_EDGES + e];

    const float f0  = edge_attr[e * 4 + 0];
    const float f1a = edge_attr[e * 4 + 1];
    const float f1b = edge_attr[e * 4 + 2];
    const float f1c = edge_attr[e * 4 + 3];

    const float* xr = x + (size_t)src * 128;
    const float x0u = xr[u];
    const float x1a = xr[32 + 3 * u + 0];
    const float x1b = xr[32 + 3 * u + 1];
    const float x1c = xr[32 + 3 * u + 2];

    const float* we = wb + el * 128;
    const float w0 = we[u];
    const float w1 = we[32 + u];
    const float w2 = we[64 + u];
    const float w3 = we[96 + u];

    const float dot = x1a * f1a + x1b * f1b + x1c * f1c;
    float* ob = out + (size_t)dst * 256;

    atomic_add_f32(ob + u,        w0 * x0u * f0 * inv_den);            // m_s0
    atomic_add_f32(ob + 32 + u,   INV_SQRT3 * w3 * dot * inv_den);     // m_s1
    const float wv0 = w1 * x0u * inv_den;                              // m_v0
    atomic_add_f32(ob + 64  + 3 * u + 0, wv0 * f1a);
    atomic_add_f32(ob + 64  + 3 * u + 1, wv0 * f1b);
    atomic_add_f32(ob + 64  + 3 * u + 2, wv0 * f1c);
    const float wv1 = w2 * f0 * inv_den;                               // m_v1
    atomic_add_f32(ob + 160 + 3 * u + 0, wv1 * x1a);
    atomic_add_f32(ob + 160 + 3 * u + 1, wv1 * x1b);
    atomic_add_f32(ob + 160 + 3 * u + 2, wv1 * x1c);
  }
}

// ---------------------------------------------------------------------------
// Host: exact SSP normalization constant (same trapz as the reference).
// ---------------------------------------------------------------------------
static float compute_ssp_c() {
  const int    N  = 48001;
  const double dz = 24.0 / (N - 1);
  const double log2c = 0.6931471805599453;
  const double inv_sqrt2pi = 0.3989422804014327;
  double integ = 0.0, gprev = 0.0;
  for (int i = 0; i < N; ++i) {
    double z  = -12.0 + dz * i;
    double la = (z > 0.0) ? z + log1p(exp(-z)) : log1p(exp(z));
    double s  = la - log2c;
    double g  = s * s * exp(-0.5 * z * z) * inv_sqrt2pi;
    if (i) integ += 0.5 * (g + gprev) * dz;
    gprev = g;
  }
  return (float)(1.0 / sqrt(integ));
}

extern "C" void kernel_launch(void* const* d_in, const int* in_sizes, int n_in,
                              void* d_out, int out_size, void* d_ws, size_t ws_size,
                              hipStream_t stream) {
  (void)in_sizes; (void)n_in; (void)ws_size;
  const float* x         = (const float*)d_in[0];
  const float* edge_attr = (const float*)d_in[1];
  const float* edge_emb  = (const float*)d_in[2];
  const int*   edge_idx  = (const int*)d_in[3];
  const float* W1        = (const float*)d_in[4];
  const float* W2        = (const float*)d_in[5];
  const float* W3        = (const float*)d_in[6];
  const float* denom     = (const float*)d_in[7];
  float* out             = (float*)d_out;
  unsigned short* wts    = (unsigned short*)d_ws;

  const float ssp_c = compute_ssp_c();

  // Atomics accumulate into out: zero it every call (graph-capture safe).
  hipMemsetAsync(d_out, 0, (size_t)out_size * sizeof(float), stream);

  prep_weights_kernel<<<8, 256, 0, stream>>>(W1, W2, W3, wts);

  // 640000 edges / (8 waves * 16 edges) = 5000 blocks, all full tiles.
  edge_mlp_message_kernel<<<5000, 256, 0, stream>>>(
      x, edge_attr, edge_emb, edge_idx, wts, denom, out, ssp_c);
}
